// GNN_node_73512660238838
// MI455X (gfx1250) — matrix operand order repaired and checked by reading, compiled, not run on hardware
//
#include <hip/hip_runtime.h>
#include <hip/hip_bf16.h>

typedef __bf16 bf16_t;
typedef __attribute__((ext_vector_type(16))) __bf16 v16bf;
typedef __attribute__((ext_vector_type(8)))  __bf16 v8bf;
typedef __attribute__((ext_vector_type(8)))  float  v8f;

#define NNODES 100000
#define NEDGES 1600000
#define DOUT   128

// -------------------- node encoder: h = emb_nt[nt] + emb_ip[ip] (64 cols, stride 128)
__global__ void k_encode(const int* __restrict__ nt, const int* __restrict__ ip,
                         const float* __restrict__ ent, const float* __restrict__ eip,
                         float* __restrict__ h) {
  int idx = blockIdx.x * blockDim.x + threadIdx.x;
  if (idx >= NNODES * 64) return;
  int n = idx >> 6, c = idx & 63;
  h[(size_t)n * DOUT + c] = ent[nt[n] * 64 + c] + eip[ip[n] * 64 + c];
}

// -------------------- zero fill
__global__ void k_zero(float* __restrict__ p, long long count) {
  long long i = (long long)blockIdx.x * blockDim.x + threadIdx.x;
  long long stride = (long long)gridDim.x * blockDim.x;
  for (; i < count; i += stride) p[i] = 0.0f;
}

// -------------------- edge scatter: one wave32 per edge, fp32 atomics
__global__ void k_scatter(const int* __restrict__ src, const int* __restrict__ dst,
                          const float* __restrict__ h, float* __restrict__ agg,
                          float* __restrict__ deg, int d) {
  int e = blockIdx.x * (blockDim.x >> 5) + (threadIdx.x >> 5);
  if (e >= NEDGES) return;
  int lane = threadIdx.x & 31;
  int s = src[e], t = dst[e];
  const float* hp = h + (size_t)s * DOUT;
  float* ap = agg + (size_t)t * DOUT;
  for (int c = lane; c < d; c += 32) atomicAdd(&ap[c], hp[c]);
  if (lane == 0) atomicAdd(&deg[t], 1.0f);
}

// -------------------- A prep: A[n] = [ agg[n]/max(deg,1) | h[n] ] in bf16, row stride 2K
__global__ void k_prepA(const float* __restrict__ agg, const float* __restrict__ deg,
                        const float* __restrict__ h, bf16_t* __restrict__ A, int K) {
  int idx = blockIdx.x * blockDim.x + threadIdx.x;
  if (idx >= NNODES * K) return;
  int n = idx / K, k = idx - n * K;
  float dg = fmaxf(deg[n], 1.0f);
  bf16_t* row = A + (size_t)n * (2 * K);
  row[k]     = (bf16_t)(agg[(size_t)n * DOUT + k] / dg);
  row[K + k] = (bf16_t)(h[(size_t)n * DOUT + k]);
}

// -------------------- weight prep: WT[j][kk] = (kk<K ? Wl[kk][j] : Wr[kk-K][j]) bf16
__global__ void k_prepW(const float* __restrict__ Wl, const float* __restrict__ Wr,
                        bf16_t* __restrict__ WT, int K) {
  int K2 = 2 * K;
  int idx = blockIdx.x * blockDim.x + threadIdx.x;
  if (idx >= DOUT * K2) return;
  int j = idx / K2, kk = idx - j * K2;
  float v = (kk < K) ? Wl[(size_t)kk * DOUT + j] : Wr[(size_t)(kk - K) * DOUT + j];
  WT[idx] = (bf16_t)v;
}

// -------------------- WMMA GEMM: C[N,128] = A[N,Kc](bf16) @ W[Kc,128](bf16), fp32 acc
// W held transposed (WT[col][k]); whole WT staged into LDS with async global->LDS
// copies (ASYNCcnt path); each wave owns a 16x128 output strip (8 accumulator tiles).
__global__ void __launch_bounds__(256)
k_gemm_wmma(const bf16_t* __restrict__ A, const bf16_t* __restrict__ WT,
            float* __restrict__ C, int Kc) {
  extern __shared__ __align__(16) char smem_raw[];
  bf16_t* sW = reinterpret_cast<bf16_t*>(smem_raw);

  // async stage of WT (DOUT x Kc bf16) into LDS, 16B per lane per issue.
  // Low 32 bits of a generic pointer into the LDS aperture are the LDS byte
  // offset (ISA 10.2), which is what GLOBAL_LOAD_ASYNC_TO_LDS's VDST wants.
  {
    int chunks = (DOUT * Kc) >> 3;                  // 16-byte chunks
    unsigned ldsoff = (unsigned)(uintptr_t)smem_raw;
    unsigned long long gbase = (unsigned long long)(uintptr_t)WT;
    for (int i = threadIdx.x; i < chunks; i += blockDim.x) {
      unsigned l = ldsoff + (unsigned)i * 16u;
      unsigned long long g = gbase + (unsigned long long)i * 16ull;
      asm volatile("global_load_async_to_lds_b128 %0, %1, off"
                   :: "v"(l), "v"(g) : "memory");
    }
    asm volatile("s_wait_asynccnt 0x0" ::: "memory");
  }
  __syncthreads();

  int wave = threadIdx.x >> 5;
  int lane = threadIdx.x & 31;
  int rowbase = (blockIdx.x * 8 + wave) << 4;     // 16 rows per wave
  if (rowbase >= NNODES) return;                  // wave-uniform exit

  int hi = lane >> 4;   // lane-group: selects K half per ISA 16-bit layouts
  int lm = lane & 15;   // M (for A) / N (for B,C)

  v8f acc[8] = {};

  const bf16_t* arow = A + (size_t)(rowbase + lm) * Kc;

  for (int kb = 0; kb < Kc; kb += 32) {
    // A 16x32 fragment: VGPR0-3 -> K = kb + hi*8 + 0..7 ; VGPR4-7 -> K = kb+16 + hi*8 + 0..7
    v8bf a0 = *reinterpret_cast<const v8bf*>(arow + kb + hi * 8);
    v8bf a1 = *reinterpret_cast<const v8bf*>(arow + kb + 16 + hi * 8);
    v16bf af = __builtin_shufflevector(a0, a1, 0,1,2,3,4,5,6,7,8,9,10,11,12,13,14,15);
#pragma unroll
    for (int t = 0; t < 8; ++t) {
      // B 32x16 fragment: elems 0-7 -> K = kb + hi*8 + 0..7 ; elems 8-15 -> K = kb+16 + hi*8 + 0..7, col = t*16+lm
      const bf16_t* bp = sW + (size_t)(t * 16 + lm) * Kc + kb + hi * 8;
      v8bf b0 = *reinterpret_cast<const v8bf*>(bp);
      v8bf b1 = *reinterpret_cast<const v8bf*>(bp + 16);
      v16bf bfv = __builtin_shufflevector(b0, b1, 0,1,2,3,4,5,6,7,8,9,10,11,12,13,14,15);
      acc[t] = __builtin_amdgcn_wmma_f32_16x16x32_bf16(
          false, af, false, bfv, (short)0, acc[t], false, false);
    }
  }

  // C/D layout: VGPR r -> lanes 0-15: M=r, lanes 16-31: M=r+8; N = lane%16
#pragma unroll
  for (int t = 0; t < 8; ++t) {
    int col = t * 16 + lm;
#pragma unroll
    for (int r = 0; r < 8; ++r) {
      int row = rowbase + hi * 8 + r;
      if (row < NNODES) C[(size_t)row * DOUT + col] = acc[t][r];
    }
  }
}

// -------------------- coalesced BN partial stats.
// Grid-stride over flat lin; stride is a multiple of 128 so each thread's
// column (i & 127) is fixed -> register accumulation, then LDS ds_add_f32
// flush, then one global atomicAdd per block per column.
__global__ void __launch_bounds__(256)
k_colpartial(const float* __restrict__ lin, float* __restrict__ acc) {
  __shared__ float ssum[DOUT], ssq[DOUT];
  for (int i = threadIdx.x; i < DOUT; i += blockDim.x) { ssum[i] = 0.f; ssq[i] = 0.f; }
  __syncthreads();

  long long total  = (long long)NNODES * DOUT;
  long long i      = (long long)blockIdx.x * blockDim.x + threadIdx.x;
  long long stride = (long long)gridDim.x * blockDim.x;   // multiple of 128
  int c = (int)(i & (DOUT - 1));                          // loop-invariant column
  float s = 0.f, s2 = 0.f;
  for (; i < total; i += stride) {
    float v = lin[i];
    s += v; s2 += v * v;
  }
  atomicAdd(&ssum[c], s);
  atomicAdd(&ssq[c], s2);
  __syncthreads();

  for (int j = threadIdx.x; j < DOUT; j += blockDim.x) {
    atomicAdd(&acc[j], ssum[j]);
    atomicAdd(&acc[DOUT + j], ssq[j]);
  }
}

__global__ void k_statsfinal(const float* __restrict__ acc, float* __restrict__ mu,
                             float* __restrict__ rv) {
  int c = threadIdx.x;
  if (c >= DOUT) return;
  float m = acc[c] / (float)NNODES;
  float var = acc[DOUT + c] / (float)NNODES - m * m;
  mu[c] = m;
  rv[c] = rsqrtf(var + 1e-5f);
}

// -------------------- BN apply (+optional ReLU), float4 vectorized
__global__ void k_bn(const float4* __restrict__ lin, const float* __restrict__ mu,
                     const float* __restrict__ rv, const float* __restrict__ g,
                     const float* __restrict__ b, float4* __restrict__ out, int relu) {
  int idx = blockIdx.x * blockDim.x + threadIdx.x;
  if (idx >= (NNODES * DOUT) / 4) return;
  int c = (idx * 4) & (DOUT - 1);
  float4 v = lin[idx];
  float4 o;
  o.x = (v.x - mu[c + 0]) * rv[c + 0] * g[c + 0] + b[c + 0];
  o.y = (v.y - mu[c + 1]) * rv[c + 1] * g[c + 1] + b[c + 1];
  o.z = (v.z - mu[c + 2]) * rv[c + 2] * g[c + 2] + b[c + 2];
  o.w = (v.w - mu[c + 3]) * rv[c + 3] * g[c + 3] + b[c + 3];
  if (relu) {
    o.x = fmaxf(o.x, 0.f); o.y = fmaxf(o.y, 0.f);
    o.z = fmaxf(o.z, 0.f); o.w = fmaxf(o.w, 0.f);
  }
  out[idx] = o;
}

extern "C" void kernel_launch(void* const* d_in, const int* in_sizes, int n_in,
                              void* d_out, int out_size, void* d_ws, size_t ws_size,
                              hipStream_t stream) {
  (void)in_sizes; (void)n_in; (void)out_size; (void)ws_size;

  const int*   node_type = (const int*)d_in[0];
  const int*   nip       = (const int*)d_in[1];
  const int*   src       = (const int*)d_in[2];
  const int*   dst       = src + NEDGES;
  const float* ent       = (const float*)d_in[3];
  const float* eip       = (const float*)d_in[4];
  const float* Wl[3] = {(const float*)d_in[5],  (const float*)d_in[10], (const float*)d_in[15]};
  const float* Wr[3] = {(const float*)d_in[7],  (const float*)d_in[12], (const float*)d_in[17]};
  const float* gg[3] = {(const float*)d_in[8],  (const float*)d_in[13], (const float*)d_in[18]};
  const float* bb[3] = {(const float*)d_in[9],  (const float*)d_in[14], (const float*)d_in[19]};
  float* out = (float*)d_out;

  // workspace carving
  char* ws = (char*)d_ws;
  size_t off = 0;
  auto carve = [&](size_t bytes) -> void* {
    void* p = ws + off;
    off += (bytes + 255) & ~(size_t)255;
    return p;
  };
  float*  h     = (float*)carve((size_t)NNODES * DOUT * sizeof(float));
  float*  lin   = (float*)carve((size_t)NNODES * DOUT * sizeof(float));
  float*  agg   = (float*)carve((size_t)NNODES * DOUT * sizeof(float));
  float*  deg   = (float*)carve((size_t)NNODES * sizeof(float));
  bf16_t* Abuf  = (bf16_t*)carve((size_t)NNODES * 256 * sizeof(bf16_t));
  bf16_t* WT    = (bf16_t*)carve((size_t)DOUT * 256 * sizeof(bf16_t));
  float*  stacc = (float*)carve(2 * DOUT * sizeof(float));
  float*  mu    = (float*)carve(DOUT * sizeof(float));
  float*  rv    = (float*)carve(DOUT * sizeof(float));

  const int T = 256;
  k_encode<<<(NNODES * 64 + T - 1) / T, T, 0, stream>>>(node_type, nip, ent, eip, h);

  const int Ks[3] = {64, 128, 128};
  for (int l = 0; l < 3; ++l) {
    int K = Ks[l], Kc = 2 * K;

    k_zero<<<2048, T, 0, stream>>>(agg, (long long)NNODES * DOUT);
    k_zero<<<512,  T, 0, stream>>>(deg, (long long)NNODES);
    k_zero<<<1,    T, 0, stream>>>(stacc, 2 * DOUT);

    k_scatter<<<(NEDGES * 32 + T - 1) / T, T, 0, stream>>>(src, dst, h, agg, deg, K);

    k_prepA<<<(NNODES * K + T - 1) / T, T, 0, stream>>>(agg, deg, h, Abuf, K);
    k_prepW<<<(DOUT * Kc + T - 1) / T, T, 0, stream>>>(Wl[l], Wr[l], WT, K);

    int rowTiles = (NNODES + 15) / 16;       // 6250
    int blocks   = (rowTiles + 7) / 8;       // 782 (8 waves x 16 rows per block)
    size_t smem  = (size_t)DOUT * Kc * sizeof(bf16_t);  // 32KB / 64KB
    k_gemm_wmma<<<blocks, T, smem, stream>>>(Abuf, WT, lin, Kc);

    k_colpartial<<<512, T, 0, stream>>>(lin, stacc);
    k_statsfinal<<<1, DOUT, 0, stream>>>(stacc, mu, rv);

    float* tgt = (l == 2) ? out : h;
    k_bn<<<((NNODES * DOUT) / 4 + T - 1) / T, T, 0, stream>>>(
        (const float4*)lin, mu, rv, gg[l], bb[l], (float4*)tgt, (l < 2) ? 1 : 0);
  }
}
